// ContextLSTM_9285719294340
// MI455X (gfx1250) — compile-verified, matrix operand order checked by reading, not run on hardware
//
#include <hip/hip_runtime.h>
#include <hip/hip_bf16.h>
#include <math.h>

// ---------------------------------------------------------------------------
// ContextLSTM for MI455X (gfx1250, wave32, WMMA).
// B=128, T=512, I=H=1024, C=512, gates 4H=4096.
//
// v2: per-step K-work split across wave pairs (x-half / h-half), f32
// accumulators reduced through LDS -> 1024 waves/step, 128 WMMAs deep
// per wave instead of 256. Fused design (no xg materialization).
// ---------------------------------------------------------------------------

typedef __bf16 bf16_t;
typedef __attribute__((ext_vector_type(8)))  __bf16 bf16x8;
typedef __attribute__((ext_vector_type(16))) __bf16 v16bf;
typedef __attribute__((ext_vector_type(8)))  float  v8f;

#define B_   128
#define T_   512
#define I_   1024
#define H_   1024
#define G4H  4096
#define C_   512

__device__ __forceinline__ bf16_t to_bf16(float f) {
    unsigned u = __builtin_bit_cast(unsigned, f);
    unsigned r = u + 0x7FFFu + ((u >> 16) & 1u);     // round to nearest even
    unsigned short s = (unsigned short)(r >> 16);
    return __builtin_bit_cast(bf16_t, s);
}

__device__ __forceinline__ float sigmoidf_(float x) {
    return 1.0f / (1.0f + __expf(-x));
}

// Wave-striped 16x32 bf16 WMMA fragment: per-lane elements
// 0..7 <- p[k..k+7], 8..15 <- p[k+16..k+23]  (lane-half K skew folded into p).
__device__ __forceinline__ v16bf load_frag(const bf16_t* __restrict__ p) {
    bf16x8 lo = *(const bf16x8*)(p);
    bf16x8 hi = *(const bf16x8*)(p + 16);
    return __builtin_shufflevector(lo, hi, 0, 1, 2, 3, 4, 5, 6, 7,
                                           8, 9, 10, 11, 12, 13, 14, 15);
}

// ---------------------------------------------------------------------------
// Setup kernels
// ---------------------------------------------------------------------------

__global__ void cvt_f32_bf16(const float* __restrict__ src,
                             bf16_t* __restrict__ dst, size_t n) {
    size_t i = (size_t)blockIdx.x * blockDim.x + threadIdx.x;
    size_t stride = (size_t)gridDim.x * blockDim.x;
    for (; i < n; i += stride) dst[i] = to_bf16(src[i]);
}

__global__ void init_state(const float* __restrict__ h0,
                           const float* __restrict__ c0,
                           bf16_t* __restrict__ hbf,
                           float* __restrict__ c) {
    int i = blockIdx.x * blockDim.x + threadIdx.x;
    if (i < B_ * H_) {
        hbf[i] = to_bf16(h0[i]);
        c[i]   = c0[i];
    }
}

// ctxg[b,n] = sum_k ctx[b,k] * Wc[n,k] + bc[n] + bi[n] + bh[n]
__global__ void ctx_gates(const float* __restrict__ ctx,
                          const float* __restrict__ Wc,
                          const float* __restrict__ bc,
                          const float* __restrict__ bi,
                          const float* __restrict__ bh,
                          float* __restrict__ ctxg) {
    int idx = blockIdx.x * blockDim.x + threadIdx.x;   // over B_*G4H
    int b = idx >> 12;
    int n = idx & (G4H - 1);
    const float* cr = ctx + (size_t)b * C_;
    const float* wr = Wc + (size_t)n * C_;
    float s = bc[n] + bi[n] + bh[n];
    #pragma unroll 4
    for (int k = 0; k < C_; ++k) s = fmaf(cr[k], wr[k], s);
    ctxg[idx] = s;
}

// ---------------------------------------------------------------------------
// One recurrent step. grid = (8 m-tiles, 16 col-tiles), block = 256 (8 waves).
// Waves 0..3: x_t@Wi^T half for col sub-tile (wave&3).
// Waves 4..7: h  @Wh^T half for the same sub-tile; accumulators reduced
// through LDS, then waves 0..3 run the pointwise LSTM cell in registers.
// ---------------------------------------------------------------------------
__global__ __launch_bounds__(256) void lstm_step(
    const bf16_t* __restrict__ xbf,    // [B,T,I] bf16
    const bf16_t* __restrict__ Wi,     // [4H,I]  bf16
    const bf16_t* __restrict__ Wh,     // [4H,H]  bf16
    const float*  __restrict__ ctxg,   // [B,4H]  f32 (biases folded)
    const bf16_t* __restrict__ hsrc,   // [B,H]   bf16 (read)
    bf16_t*       __restrict__ hdst,   // [B,H]   bf16 (write)
    float*        __restrict__ cst,    // [B,H]   f32 cell state (in place)
    float*        __restrict__ out,    // [B,T,H] f32
    float*        __restrict__ hf,     // [B,H]   f32 final h
    float*        __restrict__ cf,     // [B,H]   f32 final c
    int t)
{
    __shared__ float red[4 * 4 * 8 * 32];             // [colw][gate][e][lane] 16KB

    const int lane     = threadIdx.x & 31;
    const int wv       = threadIdx.x >> 5;            // 0..7
    const int colw     = wv & 3;                      // col sub-tile
    const int part     = wv >> 2;                     // 0 = x/Wi, 1 = h/Wh
    const int m_base   = blockIdx.x * 16;             // batch rows
    const int col_base = blockIdx.y * 64 + colw * 16; // h columns
    const int l15      = lane & 15;
    const int kb       = (lane >> 4) << 3;            // 0 or 8 (lane-half K skew)

    const int m = m_base + l15;                       // A-fragment row
    const int n = col_base + l15;                     // B-fragment column (W row)

    // part 0 -> A = x_t row, B = Wi rows; part 1 -> A = h row, B = Wh rows.
    const bf16_t* aptr = part ? (hsrc + (size_t)m * H_ + kb)
                              : (xbf + ((size_t)m * T_ + t) * I_ + kb);
    const bf16_t* bptr = part ? (Wh + (size_t)n * H_ + kb)
                              : (Wi + (size_t)n * I_ + kb);

    v8f acc[4] = {};

    for (int k = 0; k < I_; k += 32) {
        v16bf a = load_frag(aptr + k);
        #pragma unroll
        for (int g = 0; g < 4; ++g) {
            v16bf b = load_frag(bptr + (size_t)g * (1024 * I_) + k);
            acc[g] = __builtin_amdgcn_wmma_f32_16x16x32_bf16(
                false, a, false, b, (short)0, acc[g], false, false);
        }
    }

    // Reduce the h-half into LDS; x-half waves pick it up and finish.
    if (part == 1) {
        float* dst = red + ((size_t)colw * 4 * 8 + 0) * 32 + lane;
        #pragma unroll
        for (int g = 0; g < 4; ++g)
            #pragma unroll
            for (int e = 0; e < 8; ++e)
                dst[(g * 8 + e) * 32] = acc[g][e];
    }
    __syncthreads();
    if (part == 1) return;

    const float* src = red + ((size_t)colw * 4 * 8) * 32 + lane;
    #pragma unroll
    for (int g = 0; g < 4; ++g)
        #pragma unroll
        for (int e = 0; e < 8; ++e)
            acc[g][e] += src[(g * 8 + e) * 32];

    // Pointwise LSTM cell.  D layout: element e -> M = e + 8*(lane>=16),
    // N = lane&15 within the wave tile.
    const int colD = col_base + l15;
    const int mhi  = (lane >> 4) << 3;
    #pragma unroll
    for (int e = 0; e < 8; ++e) {
        const int    mD  = m_base + e + mhi;
        const float* cg  = ctxg + (size_t)mD * G4H;
        const float  ig  = sigmoidf_(acc[0][e] + cg[colD]);
        const float  fg  = sigmoidf_(acc[1][e] + cg[1024 + colD]);
        const float  gg  = tanhf(acc[2][e] + cg[2048 + colD]);
        const float  og  = sigmoidf_(acc[3][e] + cg[3072 + colD]);
        const size_t hix = (size_t)mD * H_ + colD;
        const float  cn  = fg * cst[hix] + ig * gg;
        const float  hn  = og * tanhf(cn);
        cst[hix]  = cn;
        hdst[hix] = to_bf16(hn);
        out[((size_t)mD * T_ + t) * H_ + colD] = hn;
        if (t == T_ - 1) { hf[hix] = hn; cf[hix] = cn; }
    }
}

// ---------------------------------------------------------------------------
// Host launcher
// ---------------------------------------------------------------------------
extern "C" void kernel_launch(void* const* d_in, const int* in_sizes, int n_in,
                              void* d_out, int out_size, void* d_ws, size_t ws_size,
                              hipStream_t stream) {
    const float* x   = (const float*)d_in[0];   // [B,T,I]
    const float* h0  = (const float*)d_in[1];   // [B,H]
    const float* c0  = (const float*)d_in[2];   // [B,H]
    const float* ctx = (const float*)d_in[3];   // [B,C]
    const float* Wi  = (const float*)d_in[4];   // [4H,I]
    const float* bi  = (const float*)d_in[5];   // [4H]
    const float* Wh  = (const float*)d_in[6];   // [4H,H]
    const float* bh  = (const float*)d_in[7];   // [4H]
    const float* Wc  = (const float*)d_in[8];   // [4H,C]
    const float* bc  = (const float*)d_in[9];   // [4H]

    float* out = (float*)d_out;                       // [B,T,H]
    float* hf  = out + (size_t)B_ * T_ * H_;          // [B,H]
    float* cf  = hf + (size_t)B_ * H_;                // [B,H]

    // Workspace carve-out (all regions 256B aligned by construction).
    char* w = (char*)d_ws;
    bf16_t* xbf  = (bf16_t*)w;  w += (size_t)B_ * T_ * I_ * sizeof(bf16_t); // 128 MB
    bf16_t* Wibf = (bf16_t*)w;  w += (size_t)G4H * I_ * sizeof(bf16_t);     // 8 MB
    bf16_t* Whbf = (bf16_t*)w;  w += (size_t)G4H * H_ * sizeof(bf16_t);     // 8 MB
    float*  ctxg = (float*)w;   w += (size_t)B_ * G4H * sizeof(float);      // 2 MB
    bf16_t* hb0  = (bf16_t*)w;  w += (size_t)B_ * H_ * sizeof(bf16_t);
    bf16_t* hb1  = (bf16_t*)w;  w += (size_t)B_ * H_ * sizeof(bf16_t);
    float*  cbuf = (float*)w;   w += (size_t)B_ * H_ * sizeof(float);

    // 1) One-time conversions / folds.
    cvt_f32_bf16<<<8192, 256, 0, stream>>>(x, xbf, (size_t)B_ * T_ * I_);
    cvt_f32_bf16<<<2048, 256, 0, stream>>>(Wi, Wibf, (size_t)G4H * I_);
    cvt_f32_bf16<<<2048, 256, 0, stream>>>(Wh, Whbf, (size_t)G4H * H_);
    ctx_gates<<<(B_ * G4H) / 256, 256, 0, stream>>>(ctx, Wc, bc, bi, bh, ctxg);
    init_state<<<(B_ * H_) / 256, 256, 0, stream>>>(h0, c0, hb0, cbuf);

    // 2) Recurrence: 512 dependent step launches (graph-friendly),
    //    h ping-pongs between bf16 buffers.
    for (int t = 0; t < T_; ++t) {
        bf16_t* hs = (t & 1) ? hb1 : hb0;
        bf16_t* hd = (t & 1) ? hb0 : hb1;
        lstm_step<<<dim3(8, 16), 256, 0, stream>>>(
            xbf, Wibf, Whbf, ctxg, hs, hd, cbuf, out, hf, cf, t);
    }
}
